// DWFC_MSA_16020228014355
// MI455X (gfx1250) — matrix operand order repaired and checked by reading, not compiled, not run on hardware
//
#include <hip/hip_runtime.h>
#include <hip/hip_bf16.h>

typedef __attribute__((ext_vector_type(16))) _Float16 v16h;
typedef __attribute__((ext_vector_type(8)))  _Float16 v8h;
typedef __attribute__((ext_vector_type(8)))  float    v8f;

constexpr int SG   = 32;    // spatial grid 32x32
constexpr int NTOK = 576;
constexpr int CDIM = 1024;
constexpr int BSZ  = 32;
constexpr float SCALE = 0.125f;   // (1024/16)^-0.5

// ---------------------------------------------------------------- helpers
__device__ __forceinline__ v8f wmma_f16(v16h a, v16h b, v8f c) {
    return __builtin_amdgcn_wmma_f32_16x16x32_f16(
        false, a, false, b, (short)0, c, false, false);
}

// A-fragment: lane holds row M = lane%16; half h -> K = (h>=8)*16 + laneHi*8 + (h&7)
// -> two contiguous 8-half (16B) runs at offsets laneHi*8 and 16+laneHi*8.
__device__ __forceinline__ v16h ldA_global(const _Float16* rowp, int laneHi) {
    v8h lo = *(const v8h*)(rowp + laneHi * 8);
    v8h hi = *(const v8h*)(rowp + 16 + laneHi * 8);
    v16h a;
#pragma unroll
    for (int i = 0; i < 8; ++i) { a[i] = lo[i]; a[i + 8] = hi[i]; }
    return a;
}

// B-fragment: lane holds col N = lane%16; half h -> K = (h>=8)*16 + laneHi*8 + (h&7)
// -> two contiguous 8-half (16B) runs at offsets laneHi*8 and 16+laneHi*8.
__device__ __forceinline__ v16h ldB_global(const _Float16* colp, int laneHi) {
    v8h lo = *(const v8h*)(colp + laneHi * 8);
    v8h hi = *(const v8h*)(colp + 16 + laneHi * 8);
    v16h b;
#pragma unroll
    for (int i = 0; i < 8; ++i) { b[i] = lo[i]; b[i + 8] = hi[i]; }
    return b;
}

// ---------------------------------------------------------------- kernel 0
__global__ __launch_bounds__(256) void cvt_pw_kernel(const float* __restrict__ pw,
                                                     _Float16* __restrict__ pwh) {
    int i = blockIdx.x * 256 + threadIdx.x;
    pwh[i] = (_Float16)pw[i];
}

// ---------------------------------------------------------------- kernel 1
// One block per (b,n): dwconv q/k/v -> 32x32 attention via WMMA -> scrambled
// f16 scratch write (transpose(0,2,1,3).reshape layout).
__global__ __launch_bounds__(128) void attn_kernel(
    const float* __restrict__ x,
    const float* __restrict__ qw, const float* __restrict__ qb,
    const float* __restrict__ kw, const float* __restrict__ kb,
    const float* __restrict__ vw, const float* __restrict__ vb,
    _Float16* __restrict__ oh)
{
    __shared__ float    xs [SG * SG];
    __shared__ _Float16 qs [SG * SG];   // q row-major (i,j)
    __shared__ _Float16 ks [SG * SG];   // k row-major (k,j)
    __shared__ _Float16 vts[SG * SG];   // v transposed (j,k)
    __shared__ float    sa [SG * SG];   // attn logits fp32
    __shared__ _Float16 ash[SG * SG];   // softmaxed attn f16

    const int bn  = blockIdx.x;
    const int n   = bn % NTOK;
    const int b   = bn / NTOK;
    const int tid = threadIdx.x;

    const float* xp = x + (size_t)bn * (SG * SG);
    for (int t = tid; t < SG * SG; t += 128) xs[t] = xp[t];

    float qw9[9], kw9[9], vw9[9];
#pragma unroll
    for (int t = 0; t < 9; ++t) {
        qw9[t] = qw[n * 9 + t];
        kw9[t] = kw[n * 9 + t];
        vw9[t] = vw[n * 9 + t];
    }
    const float qbv = qb[n], kbv = kb[n], vbv = vb[n];
    __syncthreads();

    // depthwise 3x3 conv, SAME zero padding; 8 pixels per thread
    for (int t = tid; t < SG * SG; t += 128) {
        const int i = t >> 5, j = t & 31;
        float aq = qbv, ak = kbv, av = vbv;
#pragma unroll
        for (int di = -1; di <= 1; ++di) {
            const int ii = i + di;
            if (ii < 0 || ii >= SG) continue;
#pragma unroll
            for (int dj = -1; dj <= 1; ++dj) {
                const int jj = j + dj;
                if (jj < 0 || jj >= SG) continue;
                const float xv = xs[ii * SG + jj];
                const int   wi = (di + 1) * 3 + (dj + 1);
                aq = fmaf(xv, qw9[wi], aq);
                ak = fmaf(xv, kw9[wi], ak);
                av = fmaf(xv, vw9[wi], av);
            }
        }
        qs[t]            = (_Float16)aq;
        ks[t]            = (_Float16)ak;
        vts[j * SG + i]  = (_Float16)av;   // store v transposed
    }
    __syncthreads();

    const int wave   = tid >> 5;
    const int lane   = tid & 31;
    const int laneHi = lane >> 4;
    const int lr     = lane & 15;
    const int ti     = wave >> 1;   // output row tile of the 32x32
    const int tk     = wave & 1;    // output col tile

    // ---- logits = Q @ K^T * SCALE ----
    v16h a, bf;
    {
        const int row = ti * 16 + lr;
#pragma unroll
        for (int h = 0; h < 16; ++h) {
            const int K = ((h >> 3) << 4) + laneHi * 8 + (h & 7);
            a[h] = qs[row * SG + K];
        }
        const int col = tk * 16 + lr;           // column of K^T == row of k
#pragma unroll
        for (int h = 0; h < 16; ++h)
            bf[h] = ks[col * SG + ((h >> 3) << 4) + laneHi * 8 + (h & 7)];
    }
    v8f c = {};
    c = wmma_f16(a, bf, c);
#pragma unroll
    for (int r = 0; r < 8; ++r) {
        const int i_full = ti * 16 + laneHi * 8 + r;
        sa[i_full * SG + tk * 16 + lr] = c[r] * SCALE;
    }
    __syncthreads();

    // ---- row softmax (fp32), one row per lane of wave 0 ----
    if (tid < SG) {
        const int row = tid;
        float m = -1e30f;
#pragma unroll
        for (int j = 0; j < SG; ++j) m = fmaxf(m, sa[row * SG + j]);
        float e[SG], sum = 0.f;
#pragma unroll
        for (int j = 0; j < SG; ++j) { e[j] = __expf(sa[row * SG + j] - m); sum += e[j]; }
        const float inv = 1.f / sum;
#pragma unroll
        for (int j = 0; j < SG; ++j) ash[row * SG + j] = (_Float16)(e[j] * inv);
    }
    __syncthreads();

    // ---- O = A @ V ----
    v16h aa, bb;
    {
        const int row = ti * 16 + lr;
#pragma unroll
        for (int h = 0; h < 16; ++h) {
            const int K = ((h >> 3) << 4) + laneHi * 8 + (h & 7);
            aa[h] = ash[row * SG + K];
        }
        const int col = tk * 16 + lr;           // output column j; vts is (j,k)
#pragma unroll
        for (int h = 0; h < 16; ++h)
            bb[h] = vts[col * SG + ((h >> 3) << 4) + laneHi * 8 + (h & 7)];
    }
    v8f co = {};
    co = wmma_f16(aa, bb, co);

    // scrambled scratch write: n' = i*18 + n/32 ; c' = (n%32)*32 + j
    const int ndiv = n >> 5, nmod = n & 31;
    _Float16* obase = oh + (size_t)b * (NTOK * CDIM);
#pragma unroll
    for (int r = 0; r < 8; ++r) {
        const int i_full = ti * 16 + laneHi * 8 + r;
        const int j_full = tk * 16 + lr;
        const int nprime = i_full * 18 + ndiv;
        const int cprime = nmod * 32 + j_full;
        obase[(size_t)nprime * CDIM + cprime] = (_Float16)co[r];
    }
}

// ---------------------------------------------------------------- kernel 2
// Projection GEMM: out[r, c'] = sum_c oh[r, c] * pw[c', c] + pb[c'].
// One wave per 64x32 output tile, 4x2 register-blocked WMMA accumulators:
// per K-step 6 fragment loads feed 8 independent WMMAs (back-to-back issue).
__global__ __launch_bounds__(256) void proj_kernel(
    const _Float16* __restrict__ oh, const _Float16* __restrict__ pwh,
    const float* __restrict__ pb, float* __restrict__ out)
{
    const int tid    = threadIdx.x;
    const int lane   = tid & 31;
    const int laneHi = lane >> 4;
    const int lr     = lane & 15;

    const int w     = blockIdx.x * 8 + (tid >> 5); // global wave id (0..9215)
    const int mTile = w >> 5;                      // 0..287 (18432 rows / 64)
    const int nTile = w & 31;                      // 0..31  (1024 cols / 32)
    const int mr    = mTile * 64;
    const int nc    = nTile * 32;

    const _Float16* arow[4];
#pragma unroll
    for (int m = 0; m < 4; ++m)
        arow[m] = oh + (size_t)(mr + m * 16 + lr) * CDIM;
    const _Float16* bcol0 = pwh + (size_t)(nc + lr)      * CDIM;  // row c' of pw = col of B
    const _Float16* bcol1 = pwh + (size_t)(nc + 16 + lr) * CDIM;

    v8f acc[4][2];
#pragma unroll
    for (int m = 0; m < 4; ++m) { acc[m][0] = (v8f){}; acc[m][1] = (v8f){}; }

#pragma unroll 2
    for (int kk = 0; kk < CDIM; kk += 32) {
        // prefetch A rows ~4 iterations ahead (A streams from HBM; B is L2-resident)
        if (kk + 128 < CDIM) {
#pragma unroll
            for (int m = 0; m < 4; ++m)
                __builtin_prefetch(arow[m] + kk + 128, 0, 0);
        }
        v16h af[4];
#pragma unroll
        for (int m = 0; m < 4; ++m) af[m] = ldA_global(arow[m] + kk, laneHi);
        const v16h b0 = ldB_global(bcol0 + kk, laneHi);
        const v16h b1 = ldB_global(bcol1 + kk, laneHi);
#pragma unroll
        for (int m = 0; m < 4; ++m) {
            acc[m][0] = wmma_f16(af[m], b0, acc[m][0]);
            acc[m][1] = wmma_f16(af[m], b1, acc[m][1]);
        }
    }

    const float bias0 = pb[nc + lr];
    const float bias1 = pb[nc + 16 + lr];
#pragma unroll
    for (int m = 0; m < 4; ++m) {
#pragma unroll
        for (int r = 0; r < 8; ++r) {
            const int row = mr + m * 16 + laneHi * 8 + r;
            out[(size_t)row * CDIM + nc + lr]      = acc[m][0][r] + bias0;
            out[(size_t)row * CDIM + nc + 16 + lr] = acc[m][1][r] + bias1;
        }
    }
}

// ---------------------------------------------------------------- launch
extern "C" void kernel_launch(void* const* d_in, const int* in_sizes, int n_in,
                              void* d_out, int out_size, void* d_ws, size_t ws_size,
                              hipStream_t stream) {
    const float* x  = (const float*)d_in[0];
    const float* qw = (const float*)d_in[1];
    const float* qb = (const float*)d_in[2];
    const float* kw = (const float*)d_in[3];
    const float* kb = (const float*)d_in[4];
    const float* vw = (const float*)d_in[5];
    const float* vb = (const float*)d_in[6];
    const float* pw = (const float*)d_in[7];
    const float* pb = (const float*)d_in[8];
    float* out = (float*)d_out;

    // workspace: [ oh: B*N*C f16 | pwh: C*C f16 ]
    _Float16* oh  = reinterpret_cast<_Float16*>(d_ws);
    _Float16* pwh = reinterpret_cast<_Float16*>(
        (char*)d_ws + (size_t)BSZ * NTOK * CDIM * sizeof(_Float16));

    cvt_pw_kernel<<<(CDIM * CDIM) / 256, 256, 0, stream>>>(pw, pwh);
    attn_kernel<<<BSZ * NTOK, 128, 0, stream>>>(x, qw, qb, kw, kb, vw, vb, oh);

    // proj: 288 mTiles(64) * 32 nTiles(32) = 9216 waves / 8 per block = 1152 blocks
    proj_kernel<<<1152, 256, 0, stream>>>(oh, pwh, pb, out);
}
